// CAML_49735721288066
// MI455X (gfx1250) — matrix operand order repaired and checked by reading, not compiled, not run on hardware
//
#include <hip/hip_runtime.h>
#include <hip/hip_bf16.h>

typedef __attribute__((ext_vector_type(16))) _Float16     v16h;
typedef __attribute__((ext_vector_type(8)))  float        v8f;
typedef __attribute__((ext_vector_type(4)))  unsigned int u32x4;

#define S_LEN   2000
#define S_PAD   2016      // multiple of 32 for GEMM2 K-chunks
#define EMB_D   100
#define HID_D   50
#define H_PAD   64        // HID padded to 2 x K=32 WMMA steps
#define KW      9
#define BATCH_N 8
#define L_OUT   8921
#define L_TILES 558       // ceil(8921/16)
#define L_PAD   (L_TILES * 16)   // 8928
#define T_TILES 63        // ceil(2000/32) conv tiles

union Frag16 { v16h h; u32x4 u[2]; };

static __device__ __forceinline__ v8f wmma_f16(v16h a, v16h b, v8f c) {
  // D = A(16x32 f16) * B(32x16 f16) + C(16x16 f32)
  return __builtin_amdgcn_wmma_f32_16x16x32_f16(false, a, false, b,
                                                (short)0, c, false, false);
}

// ---------------------------------------------------------------------------
// Kernel 0: convert u_w (L x 50 f32) -> u16 (L_PAD x 64 f16), zero padded.
// ---------------------------------------------------------------------------
__global__ __launch_bounds__(256) void cvt_uw(const float* __restrict__ uw,
                                              _Float16* __restrict__ u16) {
  int i = blockIdx.x * 256 + threadIdx.x;
  if (i >= L_PAD * H_PAD) return;
  int l = i >> 6, hd = i & 63;
  float v = (l < L_OUT && hd < HID_D) ? uw[l * HID_D + hd] : 0.0f;
  u16[i] = (_Float16)v;
}

// ---------------------------------------------------------------------------
// Kernel 1: embedding lookup + conv1d(K=9, same) + tanh.
// Writes h16  [B][S_LEN][H_PAD]   (B-operand of GEMM1, row = s)
//        hT16 [B][H_PAD][S_PAD]   (B-operand of GEMM2, row = hidden dim)
// Zero-fills all padding (hd in [50,64), s in [2000,2016)).
// ---------------------------------------------------------------------------
__global__ __launch_bounds__(256) void emb_conv(
    const int* __restrict__ x, const float* __restrict__ emb,
    const float* __restrict__ w, const float* __restrict__ cb,
    _Float16* __restrict__ h16, _Float16* __restrict__ hT16) {
  __shared__ float se[32 + KW - 1][EMB_D];  // 40 x 100 floats = 16 KB
  int b  = blockIdx.x / T_TILES;
  int t0 = (blockIdx.x % T_TILES) * 32;

  // Stage embeddings for positions [t0-4, t0+36) (zero outside sequence).
  for (int i = threadIdx.x; i < 40 * EMB_D; i += 256) {
    int p = i / EMB_D, e = i - p * EMB_D;
    int t = t0 - (KW / 2) + p;
    float v = 0.0f;
    if (t >= 0 && t < S_LEN) {
      int tok = x[b * S_LEN + t];
      v = emb[(size_t)tok * EMB_D + e];   // padding_idx row already zero
    }
    se[p][e] = v;
  }
  __syncthreads();

  for (int i = threadIdx.x; i < 32 * H_PAD; i += 256) {
    int tl = i >> 6, c = i & 63;
    int t = t0 + tl;
    float val = 0.0f;
    if (c < HID_D && t < S_LEN) {
      float acc = cb[c];
      for (int k = 0; k < KW; ++k) {
        const float* wr = w + ((size_t)c * EMB_D) * KW + k;  // w[c][e][k]
        const float* er = se[tl + k];
#pragma unroll 4
        for (int e = 0; e < EMB_D; ++e)
          acc += er[e] * wr[(size_t)e * KW];
      }
      val = tanhf(acc);
    }
    if (t < S_LEN)
      h16[((size_t)b * S_LEN + t) * H_PAD + c] = (_Float16)val;
    // t can reach t0+31 <= 2015 = S_PAD-1; write zeros into the s-padding.
    hT16[((size_t)b * H_PAD + c) * S_PAD + t] =
        (_Float16)((t < S_LEN) ? val : 0.0f);
  }
}

// ---------------------------------------------------------------------------
// Kernel 2: fused label-wise attention. One wave per (batch, 16-label tile).
// Scores strip (16 x 2016 f16 = 63 KB) lives in LDS. Both GEMM loops are
// software-pipelined (prefetch next pair of chunks before computing current).
// ---------------------------------------------------------------------------
__global__ __launch_bounds__(32) void attn(
    const _Float16* __restrict__ u16, const _Float16* __restrict__ h16,
    const _Float16* __restrict__ hT16, const float* __restrict__ out_w,
    const float* __restrict__ out_b, float* __restrict__ y) {
  __shared__ struct __align__(16) {
    _Float16 sc[16][S_PAD];   // 64512 B
    float    rsum[16];
  } sm;

  int b    = blockIdx.x / L_TILES;
  int lt   = blockIdx.x % L_TILES;
  int l0   = lt * 16;
  int lane = threadIdx.x;
  int hf   = lane >> 4;     // which 16-lane half
  int lm   = lane & 15;
  int koff = hf * 8;        // unused; A fragments also use boff
  int boff = hf * 16;       // B-fragment K offset (in f16 elements)

  // --- A fragments for GEMM1: u_w rows l0..l0+15, K = hidden dims ---
  Frag16 a0, a1;
  {
    const _Float16* ur = u16 + (size_t)(l0 + lm) * H_PAD;
    a0.u[0] = *(const u32x4*)(ur + boff);
    a0.u[1] = *(const u32x4*)(ur + boff + 8);
    a1.u[0] = *(const u32x4*)(ur + 32 + boff);
    a1.u[1] = *(const u32x4*)(ur + 32 + boff + 8);
  }

  // --- Phase 1: GEMM1, scores strip -> LDS as f16 (pipelined, chunk pairs).
  // N-loop padded to 126 chunks; chunk 125 covers s in [2000,2016): its loads
  // stay inside the workspace and its LDS stores hit the pad region, which
  // phase 2 re-zeroes. Prefetch of chunks 126/127 also stays inside ws.
  {
    const _Float16* hb = h16 + (size_t)b * S_LEN * H_PAD + (size_t)lm * H_PAD;
#define LOADB1(dst0, dst1, n)                                                \
  {                                                                          \
    const _Float16* hr = hb + (size_t)(n) * (16 * H_PAD);                    \
    dst0.u[0] = *(const u32x4*)(hr + boff);                                  \
    dst0.u[1] = *(const u32x4*)(hr + boff + 8);                              \
    dst1.u[0] = *(const u32x4*)(hr + 32 + boff);                             \
    dst1.u[1] = *(const u32x4*)(hr + 32 + boff + 8);                         \
  }
    Frag16 p0, p1, q0, q1;
    LOADB1(p0, p1, 0);
    LOADB1(q0, q1, 1);
    for (int n = 0; n < 126; n += 2) {
      Frag16 r0, r1, s0, s1;
      LOADB1(r0, r1, n + 2);          // prefetch next pair
      LOADB1(s0, s1, n + 3);
      v8f cp = {};
      cp = wmma_f16(a0.h, p0.h, cp);
      cp = wmma_f16(a1.h, p1.h, cp);
      v8f cq = {};
      cq = wmma_f16(a0.h, q0.h, cq);
      cq = wmma_f16(a1.h, q1.h, cq);
      int s = n * 16 + lm;
#pragma unroll
      for (int r = 0; r < 8; ++r) {   // C: (M = r + 8*hf, N = lm)
        sm.sc[r + 8 * hf][s]      = (_Float16)cp[r];
        sm.sc[r + 8 * hf][s + 16] = (_Float16)cq[r];
      }
      p0 = r0; p1 = r1; q0 = s0; q1 = s1;
    }
#undef LOADB1
  }
  __syncthreads();

  // --- Phase 2: unnormalized softmax, two rows in flight (one per 16-lane
  // half-wave); reductions use shfl_xor masks <= 8 (stay within the half).
  for (int i = 0; i < 8; ++i) {
    int r = 2 * i + hf;
    float mx = -3.402823466e38f;
    for (int s = lm; s < S_LEN; s += 16)
      mx = fmaxf(mx, (float)sm.sc[r][s]);
    for (int o = 8; o > 0; o >>= 1)
      mx = fmaxf(mx, __shfl_xor(mx, o, 32));
    float sum = 0.0f;
    for (int s = lm; s < S_LEN; s += 16) {
      float e = __expf((float)sm.sc[r][s] - mx);
      sm.sc[r][s] = (_Float16)e;
      sum += e;
    }
    for (int o = 8; o > 0; o >>= 1)
      sum += __shfl_xor(sum, o, 32);
    if (lm == 0) sm.rsum[r] = sum;
    sm.sc[r][S_LEN + lm] = (_Float16)0.0f;  // zero K-padding s in [2000,2016)
  }
  __syncthreads();

  // --- Phase 3: GEMM2, m_unnorm = alpha_unnorm * h (pipelined) ---
  v8f acc0 = {}, acc1 = {}, acc2 = {}, acc3 = {};
  {
    const _Float16* hTb =
        hT16 + (size_t)b * H_PAD * S_PAD + (size_t)lm * S_PAD;
#define LOADA2(da, k)                                                        \
  {                                                                          \
    da.u[0] = *(const u32x4*)&sm.sc[lm][(k) * 32 + boff];                    \
    da.u[1] = *(const u32x4*)&sm.sc[lm][(k) * 32 + boff + 8];                \
  }
#define LOADB2(d0, d1, d2, d3, k)                                            \
  {                                                                          \
    const _Float16* p = hTb + (k) * 32 + boff;                               \
    d0.u[0] = *(const u32x4*)(p);                                            \
    d0.u[1] = *(const u32x4*)(p + 8);                                        \
    d1.u[0] = *(const u32x4*)(p + 16 * S_PAD);                               \
    d1.u[1] = *(const u32x4*)(p + 16 * S_PAD + 8);                           \
    d2.u[0] = *(const u32x4*)(p + 32 * S_PAD);                               \
    d2.u[1] = *(const u32x4*)(p + 32 * S_PAD + 8);                           \
    d3.u[0] = *(const u32x4*)(p + 48 * S_PAD);                               \
    d3.u[1] = *(const u32x4*)(p + 48 * S_PAD + 8);                           \
  }
    Frag16 pa, pb0, pb1, pb2, pb3;
    LOADA2(pa, 0);
    LOADB2(pb0, pb1, pb2, pb3, 0);
    for (int k = 0; k < 63; ++k) {
      Frag16 na, nb0, nb1, nb2, nb3;
      int kn = (k < 62) ? (k + 1) : 62;   // clamp: keep prefetch in bounds
      LOADA2(na, kn);
      LOADB2(nb0, nb1, nb2, nb3, kn);
      acc0 = wmma_f16(pa.h, pb0.h, acc0);   // hd  0..15
      acc1 = wmma_f16(pa.h, pb1.h, acc1);   // hd 16..31
      acc2 = wmma_f16(pa.h, pb2.h, acc2);   // hd 32..47
      acc3 = wmma_f16(pa.h, pb3.h, acc3);   // hd 48..63 (zero rows >= 50)
      pa = na; pb0 = nb0; pb1 = nb1; pb2 = nb2; pb3 = nb3;
    }
#undef LOADA2
#undef LOADB2
  }

  // --- Phase 4: y[b,l] = out_w[l] . m[l] / Z[l] + out_b[l] ---
#pragma unroll
  for (int r = 0; r < 8; ++r) {
    int l = l0 + r + 8 * hf;                // uniform within each 16-lane half
    float p = 0.0f;
    if (l < L_OUT) {
      const float* owr = out_w + (size_t)l * HID_D;
      p += acc0[r] * owr[lm];               // hd = lm
      p += acc1[r] * owr[16 + lm];          // hd = 16+lm
      p += acc2[r] * owr[32 + lm];          // hd = 32+lm
      if (lm < 2) p += acc3[r] * owr[48 + lm];  // hd = 48,49 only
    }
    for (int o = 8; o > 0; o >>= 1)         // reduce within 16-lane half
      p += __shfl_xor(p, o, 32);
    if (lm == 0 && l < L_OUT)
      y[(size_t)b * L_OUT + l] = p / sm.rsum[r + 8 * hf] + out_b[l];
  }
}

// ---------------------------------------------------------------------------
extern "C" void kernel_launch(void* const* d_in, const int* in_sizes, int n_in,
                              void* d_out, int out_size, void* d_ws,
                              size_t ws_size, hipStream_t stream) {
  (void)in_sizes; (void)n_in; (void)out_size; (void)ws_size;
  const int*   x     = (const int*)d_in[0];
  const float* emb   = (const float*)d_in[1];
  const float* convw = (const float*)d_in[2];
  const float* convb = (const float*)d_in[3];
  const float* uw    = (const float*)d_in[4];
  const float* outw  = (const float*)d_in[5];
  const float* outb  = (const float*)d_in[6];
  float*       y     = (float*)d_out;

  char* ws = (char*)d_ws;
  const size_t u16_bytes = (size_t)L_PAD * H_PAD * 2;            // 1,142,784
  const size_t h16_bytes = (size_t)BATCH_N * S_LEN * H_PAD * 2;  // 2,048,000
  _Float16* u16  = (_Float16*)ws;
  _Float16* h16  = (_Float16*)(ws + u16_bytes);
  _Float16* hT16 = (_Float16*)(ws + u16_bytes + h16_bytes);      // +2,064,384

  cvt_uw<<<(L_PAD * H_PAD + 255) / 256, 256, 0, stream>>>(uw, u16);
  emb_conv<<<BATCH_N * T_TILES, 256, 0, stream>>>(x, emb, convw, convb,
                                                  h16, hT16);
  attn<<<BATCH_N * L_TILES, 32, 0, stream>>>(u16, h16, hT16, outw, outb, y);
}
